// GPT_72868415144566
// MI455X (gfx1250) — matrix operand order repaired and checked by reading, not compiled, not run on hardware
//
#include <hip/hip_runtime.h>

// ---------------------------------------------------------------------------
// Types for CDNA5 WMMA (wave32, 16x16x32 bf16 -> f32)
// ---------------------------------------------------------------------------
typedef __attribute__((ext_vector_type(16))) __bf16 bf16x16;
typedef __attribute__((ext_vector_type(8)))  float  f32x8;

// Problem constants (match reference exactly)
#define B_ 2
#define S_ 2048
#define D_ 2048
#define H_ 16
#define KV_ 4
#define HD_ 128
#define R_ 4
#define MLP_ 8192
#define M_ (B_ * S_)          // 4096 rows
#define KVD_ (KV_ * HD_)      // 512
#define RMS_EPS 1.1920929e-07f

// ---------------------------------------------------------------------------
// Small helpers
// ---------------------------------------------------------------------------
__device__ __forceinline__ __bf16 f2b(float f) {
  union { float f; unsigned u; } i; i.f = f;
  unsigned r = i.u + 0x7FFFu + ((i.u >> 16) & 1u);  // round-to-nearest-even
  union { unsigned short u; __bf16 b; } o; o.u = (unsigned short)(r >> 16);
  return o.b;
}
__device__ __forceinline__ float b2f(__bf16 b) {
  union { __bf16 b; unsigned short u; } i; i.b = b;
  union { unsigned u; float f; } o; o.u = ((unsigned)i.u) << 16;
  return o.f;
}
// Load 16 bf16 as two 16-byte chunks at p0 / p1 (must be 16B aligned).
__device__ __forceinline__ bf16x16 ld16(const __bf16* p0, const __bf16* p1) {
  union { uint4 u[2]; bf16x16 v; } r;
  r.u[0] = *(const uint4*)p0;
  r.u[1] = *(const uint4*)p1;
  return r.v;
}
// A-operand fragment (16x32, 16-bit): lane l holds row M=(l&15);
// K chunks {0..7,16..23} for lanes 0-15, {8..15,24..31} for lanes 16-31.
__device__ __forceinline__ bf16x16 ld_afrag(const __bf16* rowbase, int k, int lane) {
  const __bf16* p = rowbase + k + ((lane >= 16) ? 8 : 0);
  return ld16(p, p + 16);
}

// ---------------------------------------------------------------------------
// fp32 -> bf16 bulk convert (weights)
// ---------------------------------------------------------------------------
__global__ __launch_bounds__(256) void cvt_f32_bf16(const float* __restrict__ src,
                                                    __bf16* __restrict__ dst, long n) {
  long i = (long)blockIdx.x * blockDim.x + threadIdx.x;
  long st = (long)gridDim.x * blockDim.x;
  for (; i < n; i += st) dst[i] = f2b(src[i]);
}

// ---------------------------------------------------------------------------
// xm = rm0*x + rm1*x0 ; h = rms(xm)  (one block per row)
// ---------------------------------------------------------------------------
__global__ __launch_bounds__(256) void premix_rms_kernel(
    const float* __restrict__ x, const float* __restrict__ x0,
    const float* __restrict__ rmix, float* __restrict__ xm,
    __bf16* __restrict__ hbf) {
  long row = blockIdx.x;
  const float* px  = x  + row * D_;
  const float* px0 = x0 + row * D_;
  float* pxm = xm + row * D_;
  __bf16* ph = hbf + row * D_;
  __shared__ float red[8];
  float ss = 0.f;
  for (int d = threadIdx.x; d < D_; d += 256) {
    float v = rmix[d] * px[d] + rmix[D_ + d] * px0[d];
    pxm[d] = v;
    ss += v * v;
  }
  for (int o = 16; o > 0; o >>= 1) ss += __shfl_xor(ss, o);
  if ((threadIdx.x & 31) == 0) red[threadIdx.x >> 5] = ss;
  __syncthreads();
  float tot = 0.f;
#pragma unroll
  for (int i = 0; i < 8; i++) tot += red[i];
  float inv = rsqrtf(tot / (float)D_ + RMS_EPS);
  for (int d = threadIdx.x; d < D_; d += 256) ph[d] = f2b(pxm[d] * inv);
}

// rms of fp32 input -> bf16 output (one block per row, width D_)
__global__ __launch_bounds__(256) void rms_bf_kernel(const float* __restrict__ in,
                                                     __bf16* __restrict__ out) {
  long row = blockIdx.x;
  const float* p = in + row * D_;
  __bf16* q = out + row * D_;
  __shared__ float red[8];
  float ss = 0.f;
  for (int d = threadIdx.x; d < D_; d += 256) { float v = p[d]; ss += v * v; }
  for (int o = 16; o > 0; o >>= 1) ss += __shfl_xor(ss, o);
  if ((threadIdx.x & 31) == 0) red[threadIdx.x >> 5] = ss;
  __syncthreads();
  float tot = 0.f;
#pragma unroll
  for (int i = 0; i < 8; i++) tot += red[i];
  float inv = rsqrtf(tot / (float)D_ + RMS_EPS);
  for (int d = threadIdx.x; d < D_; d += 256) q[d] = f2b(p[d] * inv);
}

// ---------------------------------------------------------------------------
// Generic TN GEMM:  C[M,N] = A[M,K] (bf16, row-major) x W[N,K]^T (bf16, row-major)
// Workgroup = 4 waves; block computes 256(M) x 64(N). All 4 waves share one
// 64x32 W tile staged in LDS via gfx1250 async LDS loads (ASYNCcnt-tracked,
// double buffered: s_wait_asynccnt 0x2 waits only on the previous batch).
// Per wave: 16 v_wmma_f32_16x16x32_bf16 per 32-wide K step.
// Epilogue modes: 0 = store f32 (+optional bias)
//                 2 = leaky-square activation -> bf16 (MLP up)
//                 3 = outF = base + scale[col]*acc   (fused residual)
// ---------------------------------------------------------------------------
__global__ __launch_bounds__(128) void gemm_bf16_tn(
    const __bf16* __restrict__ A, const __bf16* __restrict__ W,
    float* __restrict__ outF, __bf16* __restrict__ outB,
    const float* __restrict__ bias, const float* __restrict__ base,
    const float* __restrict__ scale, int M, int N, int K, int mode) {
  __shared__ __attribute__((aligned(16))) __bf16 bsh[2][64 * 32];  // 8 KB
  const int tid  = threadIdx.x;
  const int lane = tid & 31;
  const int wave = tid >> 5;
  const int m0 = (blockIdx.x * 4 + wave) * 64;
  const int n0 = blockIdx.y * 64;

  const unsigned sbase = (unsigned)(size_t)(&bsh[0][0]);
  // Stage the 64x32 bf16 W tile for K offset kk into bsh[buf].
  // 256 x 16B chunks, 128 threads -> 2 async b128 per thread.
  auto stage = [&](int buf, int kk) {
#pragma unroll
    for (int p = 0; p < 2; p++) {
      int cc = tid + p * 128;      // chunk id 0..255
      int n  = cc >> 2;            // tile row (0..63)
      int c  = cc & 3;             // 16B chunk within row
      unsigned loff = sbase + (unsigned)(buf * 4096 + n * 64 + c * 16);
      const __bf16* g = W + (size_t)(n0 + n) * K + kk + c * 8;
      asm volatile("global_load_async_to_lds_b128 %0, %1, off"
                   :: "v"(loff), "v"((const void*)g) : "memory");
    }
  };

  f32x8 acc[4][4] = {};
  const __bf16* arow[4];
#pragma unroll
  for (int i = 0; i < 4; i++) arow[i] = A + (long)(m0 + 16 * i + (lane & 15)) * K;

  stage(0, 0);
  const int nk = K / 32;
  for (int t = 0; t < nk; t++) {
    const int k = t * 32;
    const int buf = t & 1;
    if (t + 1 < nk) {
      stage(buf ^ 1, k + 32);                              // prefetch next tile
      asm volatile("s_wait_asynccnt 0x2" ::: "memory");    // wait prev batch only
    } else {
      asm volatile("s_wait_asynccnt 0x0" ::: "memory");
    }
    __syncthreads();  // staged tile visible to all 4 waves

    bf16x16 a[4], b[4];
#pragma unroll
    for (int i = 0; i < 4; i++) a[i] = ld_afrag(arow[i], k, lane);
#pragma unroll
    for (int j = 0; j < 4; j++) {
      const __bf16* p =
          &bsh[buf][(16 * j + (lane & 15)) * 32 + ((lane >= 16) ? 8 : 0)];
      b[j] = ld16(p, p + 16);
    }
#pragma unroll
    for (int i = 0; i < 4; i++) {
      __builtin_prefetch(arow[i] + k + 32, 0, 3);          // global_prefetch_b8
#pragma unroll
      for (int j = 0; j < 4; j++)
        acc[i][j] = __builtin_amdgcn_wmma_f32_16x16x32_bf16(
            false, a[i], false, b[j], (short)0, acc[i][j], false, false);
    }
    __syncthreads();  // all reads of bsh[buf] done before it is re-staged
  }

  const int colL = lane & 15;
  const int rowH = (lane >= 16) ? 8 : 0;
#pragma unroll
  for (int i = 0; i < 4; i++) {
#pragma unroll
    for (int j = 0; j < 4; j++) {
      int cb = n0 + 16 * j + colL;
#pragma unroll
      for (int e = 0; e < 8; e++) {
        int r = m0 + 16 * i + e + rowH;
        long idx = (long)r * N + cb;
        float va = acc[i][j][e];
        if (bias) va += bias[idx];
        if (mode == 0) {
          outF[idx] = va;
        } else if (mode == 2) {
          float t2 = (va >= 0.f) ? va : 0.5f * va;  // leaky
          outB[idx] = f2b(t2 * t2);                 // squared
        } else {
          outF[idx] = base[idx] + scale[cb] * va;
        }
      }
    }
  }
}

// ---------------------------------------------------------------------------
// gate = sigmoid(h @ attn_gate_w^T + attn_gate_b)  (one block per row)
// ---------------------------------------------------------------------------
__global__ __launch_bounds__(256) void gate_kernel(
    const __bf16* __restrict__ h, const float* __restrict__ gw,
    const float* __restrict__ gb, float* __restrict__ gate) {
  long row = blockIdx.x;
  int wave = threadIdx.x >> 5, lane = threadIdx.x & 31;
  const __bf16* ph = h + row * D_;
  for (int hh = wave; hh < H_; hh += 8) {
    const float* pw = gw + (long)hh * D_;
    float s = 0.f;
    for (int d = lane; d < D_; d += 32) s += b2f(ph[d]) * pw[d];
    for (int o = 16; o > 0; o >>= 1) s += __shfl_xor(s, o);
    if (lane == 0) gate[row * H_ + hh] = 1.f / (1.f + __expf(-(s + gb[hh])));
  }
}

// ---------------------------------------------------------------------------
// QKV post-process (one block of 128 threads per (b,s) row; 4 waves).
//  q: per-head rms, NTK rope, q_gain -> bf16
//  k: per-head rms, rope            -> bf16
//  v: v_resid out (fp32), lambda-mix with v0 -> fp32 + transposed bf16 [b,kv,d,s]
// ---------------------------------------------------------------------------
__global__ __launch_bounds__(128) void qkv_post_kernel(
    const float* __restrict__ qraw, const float* __restrict__ kraw,
    const float* __restrict__ vraw, const float* __restrict__ v0,
    const float* __restrict__ qgain, const float* __restrict__ vrl,
    __bf16* __restrict__ qbf, __bf16* __restrict__ kbf,
    __bf16* __restrict__ vbft, float* __restrict__ vmix,
    float* __restrict__ vresid) {
  long row = blockIdx.x;
  int b = (int)(row / S_), s = (int)(row % S_);
  int wave = threadIdx.x >> 5, lane = threadIdx.x & 31;
  // NTK-scaled rope base: S=2048 > TRAIN=1024 -> base = 10000 * 2^(128/126)
  const float lbase = logf(10000.f) + (128.f / 126.f) * logf(2.f);
  float fr0 = __expf(-((float)lane / 64.f) * lbase);          // j = lane
  float fr1 = __expf(-((float)(lane + 32) / 64.f) * lbase);   // j = lane+32
  float a0 = (float)s * fr0, a1 = (float)s * fr1;
  float c0 = cosf(a0), s0 = sinf(a0), c1 = cosf(a1), s1 = sinf(a1);

  // ---- Q heads (wave handles heads wave, wave+4, wave+8, wave+12) ----
  for (int h = wave; h < H_; h += 4) {
    const float* p = qraw + row * D_ + h * HD_;
    float x0 = p[lane], x1 = p[lane + 32], x2 = p[lane + 64], x3 = p[lane + 96];
    float ss = x0 * x0 + x1 * x1 + x2 * x2 + x3 * x3;
    for (int o = 16; o > 0; o >>= 1) ss += __shfl_xor(ss, o);
    float inv = rsqrtf(ss / (float)HD_ + RMS_EPS);
    x0 *= inv; x1 *= inv; x2 *= inv; x3 *= inv;
    float g = qgain[h];
    __bf16* q = qbf + row * D_ + h * HD_;
    q[lane]      = f2b(( x0 * c0 + x2 * s0) * g);
    q[lane + 32] = f2b(( x1 * c1 + x3 * s1) * g);
    q[lane + 64] = f2b((-x0 * s0 + x2 * c0) * g);
    q[lane + 96] = f2b((-x1 * s1 + x3 * c1) * g);
  }
  // ---- K heads (wave == kv head) ----
  {
    int kv = wave;
    const float* p = kraw + row * KVD_ + kv * HD_;
    float x0 = p[lane], x1 = p[lane + 32], x2 = p[lane + 64], x3 = p[lane + 96];
    float ss = x0 * x0 + x1 * x1 + x2 * x2 + x3 * x3;
    for (int o = 16; o > 0; o >>= 1) ss += __shfl_xor(ss, o);
    float inv = rsqrtf(ss / (float)HD_ + RMS_EPS);
    x0 *= inv; x1 *= inv; x2 *= inv; x3 *= inv;
    __bf16* kq = kbf + row * KVD_ + kv * HD_;
    kq[lane]      = f2b( x0 * c0 + x2 * s0);
    kq[lane + 32] = f2b( x1 * c1 + x3 * s1);
    kq[lane + 64] = f2b(-x0 * s0 + x2 * c0);
    kq[lane + 96] = f2b(-x1 * s1 + x3 * c1);
  }
  // ---- V (wave == kv head) ----
  {
    int kv = wave;
    long base = row * KVD_ + kv * HD_;
    float l0 = vrl[0], l1 = vrl[1];
    __bf16* t = vbft + ((long)(b * KV_ + kv) * HD_) * S_;
#pragma unroll
    for (int c = 0; c < 4; c++) {
      int d = lane + 32 * c;
      float r = vraw[base + d];
      vresid[base + d] = r;                       // v_resid output
      float m = l0 * v0[base + d] + l1 * r;       // lambda mix
      vmix[base + d] = m;
      t[(long)d * S_ + s] = f2b(m);               // transposed bf16 [b,kv,d,s]
    }
  }
}

// ---------------------------------------------------------------------------
// Flash attention: one wave handles 16 query rows for one (b,h).
// QK^T and P*V via bf16 WMMA; online softmax in fp32; P transits through LDS
// to convert C-matrix layout -> A-operand layout (same-wave LDS is in-order).
// ---------------------------------------------------------------------------
__global__ __launch_bounds__(32) void attn_flash(
    const __bf16* __restrict__ qbf, const __bf16* __restrict__ kbf,
    const __bf16* __restrict__ vbft, float* __restrict__ y) {
  const int lane = threadIdx.x;
  const int q0 = blockIdx.x * 16;
  const int bh = blockIdx.y;
  const int b = bh >> 4;       // H_=16
  const int h = bh & 15;
  const int kv = h >> 2;       // R_=4
  const __bf16* Q  = qbf + (long)b * S_ * D_ + h * HD_;
  const __bf16* Kb = kbf + (long)b * S_ * KVD_ + kv * HD_;
  const __bf16* Vt = vbft + (long)(b * KV_ + kv) * HD_ * S_;

  const int colL = lane & 15;
  const int hi16 = (lane >= 16);
  const int rowH = hi16 ? 8 : 0;

  // Q fragments (16 rows x 128, 4 chunks of K=32) loaded once
  bf16x16 qa[4];
  {
    const __bf16* qr = Q + (long)(q0 + colL) * D_;
#pragma unroll
    for (int c = 0; c < 4; c++) {
      const __bf16* p = qr + c * 32 + (hi16 ? 8 : 0);
      qa[c] = ld16(p, p + 16);
    }
  }
  f32x8 acc[8] = {};
  float rowm[8], rowl[8];
#pragma unroll
  for (int e = 0; e < 8; e++) { rowm[e] = -1e30f; rowl[e] = 0.f; }

  __shared__ __attribute__((aligned(16))) __bf16 pl[16 * 32];
  const float scl = 0.088388347648318f;  // 1/sqrt(128)
  const int kend = q0 + 16;
  for (int kt = 0; kt < kend; kt += 32) {
    if (kt + 32 < kend) {
      __builtin_prefetch(Kb + (long)(kt + 32 + colL) * KVD_, 0, 3);
    }
    // scores for 32 keys (two 16-col WMMA tiles)
    f32x8 sc0 = {}, sc1 = {};
#pragma unroll
    for (int c = 0; c < 4; c++) {
      const __bf16* p = Kb + (long)(kt + colL) * KVD_ + c * 32 + (hi16 ? 8 : 0);
      bf16x16 kb = ld16(p, p + 16);
      sc0 = __builtin_amdgcn_wmma_f32_16x16x32_bf16(false, qa[c], false, kb,
                                                    (short)0, sc0, false, false);
    }
#pragma unroll
    for (int c = 0; c < 4; c++) {
      const __bf16* p = Kb + (long)(kt + 16 + colL) * KVD_ + c * 32 + (hi16 ? 8 : 0);
      bf16x16 kb = ld16(p, p + 16);
      sc1 = __builtin_amdgcn_wmma_f32_16x16x32_bf16(false, qa[c], false, kb,
                                                    (short)0, sc1, false, false);
    }
    // online softmax (row stats via 16-lane shuffles; matches C layout halves)
#pragma unroll
    for (int e = 0; e < 8; e++) {
      int r = q0 + e + rowH;
      float a0 = (kt + colL <= r)      ? sc0[e] * scl : -1e30f;
      float a1 = (kt + 16 + colL <= r) ? sc1[e] * scl : -1e30f;
      float mx = fmaxf(a0, a1);
      mx = fmaxf(mx, __shfl_xor(mx, 1));
      mx = fmaxf(mx, __shfl_xor(mx, 2));
      mx = fmaxf(mx, __shfl_xor(mx, 4));
      mx = fmaxf(mx, __shfl_xor(mx, 8));
      float nm = fmaxf(rowm[e], mx);
      float f = __expf(rowm[e] - nm);
      rowm[e] = nm;
      float p0 = __expf(a0 - nm);
      float p1 = __expf(a1 - nm);
      float ps = p0 + p1;
      ps += __shfl_xor(ps, 1); ps += __shfl_xor(ps, 2);
      ps += __shfl_xor(ps, 4); ps += __shfl_xor(ps, 8);
      rowl[e] = rowl[e] * f + ps;
#pragma unroll
      for (int j = 0; j < 8; j++) acc[j][e] *= f;
      pl[(e + rowH) * 32 + colL]      = f2b(p0);
      pl[(e + rowH) * 32 + 16 + colL] = f2b(p1);
    }
    // P (16x32) as A-operand from LDS
    bf16x16 pa;
    {
      const __bf16* pp = &pl[colL * 32 + (hi16 ? 8 : 0)];
      pa = ld16(pp, pp + 16);
    }
    // P * V : 8 d-chunks of 16 columns; Vt is [d][s] so per-lane keys contiguous
#pragma unroll
    for (int j = 0; j < 8; j++) {
      const __bf16* vp = Vt + (long)(j * 16 + colL) * S_ + kt + (hi16 ? 8 : 0);
      bf16x16 vb = ld16(vp, vp + 16);
      acc[j] = __builtin_amdgcn_wmma_f32_16x16x32_bf16(false, pa, false, vb,
                                                       (short)0, acc[j], false, false);
    }
  }
  // normalize and write y[b,s,h,d] fp32
#pragma unroll
  for (int e = 0; e < 8; e++) {
    int r = q0 + e + rowH;
    float inv = 1.f / rowl[e];
    float* yr = y + ((long)(b * S_ + r) * H_ + h) * HD_;
#pragma unroll
    for (int j = 0; j < 8; j++) yr[j * 16 + colL] = acc[j][e] * inv;
  }
}

// ---------------------------------------------------------------------------
// y <- (y - (y . vhat) vhat) * gate  -> bf16   (one block per (b,s); wave=kv)
// ---------------------------------------------------------------------------
__global__ __launch_bounds__(128) void reject_gate_kernel(
    const float* __restrict__ y, const float* __restrict__ vmix,
    const float* __restrict__ gate, __bf16* __restrict__ ybf) {
  long row = blockIdx.x;
  int wave = threadIdx.x >> 5, lane = threadIdx.x & 31;
  int kv = wave;
  const float* vp = vmix + row * KVD_ + kv * HD_;
  float v0_ = vp[lane], v1_ = vp[lane + 32], v2_ = vp[lane + 64], v3_ = vp[lane + 96];
  float ss = v0_ * v0_ + v1_ * v1_ + v2_ * v2_ + v3_ * v3_;
  for (int o = 16; o > 0; o >>= 1) ss += __shfl_xor(ss, o);
  float inv = 1.f / fmaxf(sqrtf(ss), 1e-12f);
  float n0 = v0_ * inv, n1 = v1_ * inv, n2 = v2_ * inv, n3 = v3_ * inv;
  for (int r = 0; r < R_; r++) {
    int hh = kv * R_ + r;
    const float* yp = y + (row * H_ + hh) * HD_;
    float y0 = yp[lane], y1 = yp[lane + 32], y2 = yp[lane + 64], y3 = yp[lane + 96];
    float dot = y0 * n0 + y1 * n1 + y2 * n2 + y3 * n3;
    for (int o = 16; o > 0; o >>= 1) dot += __shfl_xor(dot, o);
    float g = gate[row * H_ + hh];
    __bf16* op = ybf + row * D_ + hh * HD_;
    op[lane]      = f2b((y0 - dot * n0) * g);
    op[lane + 32] = f2b((y1 - dot * n1) * g);
    op[lane + 64] = f2b((y2 - dot * n2) * g);
    op[lane + 96] = f2b((y3 - dot * n3) * g);
  }
}

// ---------------------------------------------------------------------------
// Launch
// ---------------------------------------------------------------------------
extern "C" void kernel_launch(void* const* d_in, const int* in_sizes, int n_in,
                              void* d_out, int out_size, void* d_ws, size_t ws_size,
                              hipStream_t stream) {
  (void)in_sizes; (void)n_in; (void)out_size; (void)ws_size;
  const float* x      = (const float*)d_in[0];
  const float* x0     = (const float*)d_in[1];
  const float* q_w    = (const float*)d_in[2];
  const float* k_w    = (const float*)d_in[3];
  const float* v_w    = (const float*)d_in[4];
  const float* out_w  = (const float*)d_in[5];
  const float* up_w   = (const float*)d_in[6];
  const float* down_w = (const float*)d_in[7];
  const float* v_emb  = (const float*)d_in[8];
  const float* v0     = (const float*)d_in[9];
  const float* qgain  = (const float*)d_in[10];
  const float* agw    = (const float*)d_in[11];
  const float* agb    = (const float*)d_in[12];
  const float* vrl    = (const float*)d_in[13];
  const float* ascale = (const float*)d_in[14];
  const float* mscale = (const float*)d_in[15];
  const float* rmix   = (const float*)d_in[16];

  float* c_out = (float*)d_out;                       // (B,S,D) fp32
  float* vres  = c_out + (long)M_ * D_;               // (B,S,KV,HD) fp32

  // workspace bump allocator
  char* wsb = (char*)d_ws;
  size_t off = 0;
  auto alloc = [&](size_t bytes) -> void* {
    off = (off + 255) & ~(size_t)255;
    void* p = wsb + off;
    off += bytes;
    return p;
  };
  __bf16* wq   = (__bf16*)alloc((size_t)D_ * D_ * 2);
  __bf16* wk   = (__bf16*)alloc((size_t)KVD_ * D_ * 2);
  __bf16* wv   = (__bf16*)alloc((size_t)KVD_ * D_ * 2);
  __bf16* wo   = (__bf16*)alloc((size_t)D_ * D_ * 2);
  __bf16* wu   = (__bf16*)alloc((size_t)MLP_ * D_ * 2);
  __bf16* wd   = (__bf16*)alloc((size_t)D_ * MLP_ * 2);
  float*  xm   = (float*) alloc((size_t)M_ * D_ * 4);
  __bf16* hbf  = (__bf16*)alloc((size_t)M_ * D_ * 2);
  float*  big0 = (float*) alloc((size_t)M_ * D_ * 4);   // q f32, later y f32
  float*  kf   = (float*) alloc((size_t)M_ * KVD_ * 4);
  float*  vraw = (float*) alloc((size_t)M_ * KVD_ * 4);
  float*  vmix = (float*) alloc((size_t)M_ * KVD_ * 4);
  __bf16* qbf  = (__bf16*)alloc((size_t)M_ * D_ * 2);
  __bf16* kbf  = (__bf16*)alloc((size_t)M_ * KVD_ * 2);
  __bf16* vbft = (__bf16*)alloc((size_t)M_ * KVD_ * 2); // [b,kv,d,s]
  float*  gtb  = (float*) alloc((size_t)M_ * H_ * 4);
  __bf16* ybf  = (__bf16*)alloc((size_t)M_ * D_ * 2);
  __bf16* mbf  = (__bf16*)alloc((size_t)M_ * D_ * 2);
  __bf16* ubf  = (__bf16*)alloc((size_t)M_ * MLP_ * 2);

  // 1) convert weights to bf16 (deterministic, every launch)
  cvt_f32_bf16<<<4096, 256, 0, stream>>>(q_w,    wq, (long)D_ * D_);
  cvt_f32_bf16<<<2048, 256, 0, stream>>>(k_w,    wk, (long)KVD_ * D_);
  cvt_f32_bf16<<<2048, 256, 0, stream>>>(v_w,    wv, (long)KVD_ * D_);
  cvt_f32_bf16<<<4096, 256, 0, stream>>>(out_w,  wo, (long)D_ * D_);
  cvt_f32_bf16<<<8192, 256, 0, stream>>>(up_w,   wu, (long)MLP_ * D_);
  cvt_f32_bf16<<<8192, 256, 0, stream>>>(down_w, wd, (long)D_ * MLP_);

  // 2) residual mix + RMS
  premix_rms_kernel<<<M_, 256, 0, stream>>>(x, x0, rmix, xm, hbf);

  // 3) projections (WMMA GEMMs; 4 waves/block, async-LDS weight staging)
  gemm_bf16_tn<<<dim3(M_ / 256, D_ / 64), 128, 0, stream>>>(
      hbf, wq, big0, nullptr, nullptr, nullptr, nullptr, M_, D_, D_, 0);
  gemm_bf16_tn<<<dim3(M_ / 256, KVD_ / 64), 128, 0, stream>>>(
      hbf, wk, kf, nullptr, nullptr, nullptr, nullptr, M_, KVD_, D_, 0);
  gemm_bf16_tn<<<dim3(M_ / 256, KVD_ / 64), 128, 0, stream>>>(
      hbf, wv, vraw, nullptr, v_emb, nullptr, nullptr, M_, KVD_, D_, 0);

  // 4) attention gate (small projection) + QKV post-process
  gate_kernel<<<M_, 256, 0, stream>>>(hbf, agw, agb, gtb);
  qkv_post_kernel<<<M_, 128, 0, stream>>>(big0, kf, vraw, v0, qgain, vrl,
                                          qbf, kbf, vbft, vmix, vres);

  // 5) flash attention (y overwrites big0)
  attn_flash<<<dim3(S_ / 16, B_ * H_), 32, 0, stream>>>(qbf, kbf, vbft, big0);

  // 6) v-direction rejection + gating
  reject_gate_kernel<<<M_, 128, 0, stream>>>(big0, vmix, gtb, ybf);

  // 7) out-proj with fused residual: c = xm + attn_scale * (y @ out_w^T)
  gemm_bf16_tn<<<dim3(M_ / 256, D_ / 64), 128, 0, stream>>>(
      ybf, wo, c_out, nullptr, nullptr, xm, ascale, M_, D_, D_, 3);

  // 8) MLP: m = rms(c); u = act(m @ up_w^T); c += mlp_scale * (u @ down_w^T)
  rms_bf_kernel<<<M_, 256, 0, stream>>>(c_out, mbf);
  gemm_bf16_tn<<<dim3(M_ / 256, MLP_ / 64), 128, 0, stream>>>(
      mbf, wu, nullptr, ubf, nullptr, nullptr, nullptr, M_, MLP_, D_, 2);
  gemm_bf16_tn<<<dim3(M_ / 256, D_ / 64), 128, 0, stream>>>(
      ubf, wd, c_out, nullptr, nullptr, c_out, mscale, M_, D_, MLP_, 3);
}
